// FlashT5Attention_29721173688371
// MI455X (gfx1250) — matrix operand order repaired and checked by reading, not compiled, not run on hardware
//
#include <hip/hip_runtime.h>
#include <hip/hip_bf16.h>
#include <math.h>

typedef __bf16 bf16;
typedef __attribute__((ext_vector_type(16))) __bf16 v16bf;
typedef __attribute__((ext_vector_type(8)))  float   v8f;

#define B_  2
#define S_  2048
#define D_  1024
#define H_  16
#define DK_ 64
#define SM_SCALE 0.25f   // 1/sqrt(H)

union FragU { v16bf v; float4 f[2]; };

// A-fragment (16x32 bf16): lane holds row M=lane%16; half selects K pattern
__device__ inline v16bf load_a_frag(const bf16* row, int half) {
  FragU u;
  u.f[0] = *reinterpret_cast<const float4*>(row + half * 8);
  u.f[1] = *reinterpret_cast<const float4*>(row + 16 + half * 8);
  return u.v;
}
// B-fragment (32x16 bf16, supplied transposed: row = column n, contiguous K)
__device__ inline v16bf load_b_frag(const bf16* row, int half) {
  FragU u;
  u.f[0] = *reinterpret_cast<const float4*>(row + half * 8);
  u.f[1] = *reinterpret_cast<const float4*>(row + 16 + half * 8);
  return u.v;
}

__device__ inline v8f wmma_bf16(v16bf a, v16bf b, v8f c) {
  return __builtin_amdgcn_wmma_f32_16x16x32_bf16(false, a, false, b, (short)0, c,
                                                 false, false);
}

// ---------------- stage 0: casts / weight transpose ----------------
__global__ void cast_f32_bf16(const float* __restrict__ in, bf16* __restrict__ out, int n) {
  int i = blockIdx.x * blockDim.x + threadIdx.x;
  if (i < n) out[i] = (bf16)in[i];
}

__global__ void transpose_cast(const float* __restrict__ W, bf16* __restrict__ WT,
                               int rows, int cols) {
  int i = blockIdx.x * blockDim.x + threadIdx.x;
  if (i < rows * cols) {
    int r = i / cols, c = i % cols;
    WT[(size_t)c * rows + r] = (bf16)W[(size_t)r * cols + c];
  }
}

// ---------------- generic bf16 WMMA GEMM: C[M,N] = A[M,K] * Bt[N,K]^T -------
// Each wave: 32x64 tile -> 2 A-frags x 4 B-frags -> 8 WMMAs per 32-wide K-step.
// No software prefetch here: operands are L2-resident and extra address pairs
// push the register allocator into hot-loop spills.
// mode 0: f32 row-major   mode 1: bf16 row-major   mode 2: bf16 transposed
__global__ void gemm_wmma(const bf16* __restrict__ A, const bf16* __restrict__ Bt,
                          float* __restrict__ Cf, bf16* __restrict__ Cb,
                          int M, int N, int K, int mode) {
  int lane = threadIdx.x & 31;
  int half = lane >> 4;
  int l16  = lane & 15;
  int wave = (blockIdx.x * blockDim.x + threadIdx.x) >> 5;
  int tilesN = N >> 6;
  int mt = wave / tilesN;
  int nt = wave % tilesN;
  if (mt * 32 >= M) return;

  const bf16* arow0 = A + (size_t)(mt * 32 + l16) * K;
  const bf16* arow1 = A + (size_t)(mt * 32 + 16 + l16) * K;
  const bf16* brow[4];
#pragma unroll
  for (int j = 0; j < 4; ++j)
    brow[j] = Bt + (size_t)(nt * 64 + 16 * j + l16) * K;

  v8f acc[2][4];
#pragma unroll
  for (int i = 0; i < 2; ++i)
#pragma unroll
    for (int j = 0; j < 4; ++j) acc[i][j] = (v8f){};

  for (int k0 = 0; k0 < K; k0 += 32) {
    v16bf a0 = load_a_frag(arow0 + k0, half);
    v16bf a1 = load_a_frag(arow1 + k0, half);
#pragma unroll
    for (int j = 0; j < 4; ++j) {
      v16bf b = load_b_frag(brow[j] + k0, half);
      acc[0][j] = wmma_bf16(a0, b, acc[0][j]);
      acc[1][j] = wmma_bf16(a1, b, acc[1][j]);
    }
  }

#pragma unroll
  for (int i = 0; i < 2; ++i)
#pragma unroll
    for (int j = 0; j < 4; ++j)
#pragma unroll
      for (int r = 0; r < 8; ++r) {
        int row = mt * 32 + 16 * i + r + 8 * half;
        int col = nt * 64 + 16 * j + l16;
        if (mode == 0) {
          Cf[(size_t)row * N + col] = acc[i][j][r];
        } else if (mode == 1) {
          Cb[(size_t)row * N + col] = (bf16)acc[i][j][r];
        } else {
          Cb[(size_t)col * M + row] = (bf16)acc[i][j][r];
        }
      }
}

// ---------------- T5 relative-position bucket (NB=32 bidirectional) ---------
// max_exact=8, MAX_DIST=128 -> large = 8 + floor(2*log2(|rp|) - 6), sat at 15
__device__ inline int t5_bucket(int rp) {
  int ret = (rp > 0) ? 16 : 0;
  int rpa = rp < 0 ? -rp : rp;
  if (rpa < 8) return ret + rpa;
  int large = 8 + (int)(2.0f * __log2f((float)rpa) - 6.0f);
  if (large > 15) large = 15;
  return ret + large;
}

// ---------------- fused flash attention -------------------------------------
// One wave per (b, h, 16-row q tile). K/V streamed in 32-column chunks.
__global__ void flash_attn_t5(const bf16* __restrict__ Q, const bf16* __restrict__ Km,
                              const bf16* __restrict__ Vt,  // [D_][B_*S_] (dk-major)
                              const float* __restrict__ rel_bias,
                              bf16* __restrict__ Oout) {
  __shared__ __align__(16) bf16 pbuf[4][16][32];
  int lane = threadIdx.x & 31, half = lane >> 4, l16 = lane & 15;
  int w    = threadIdx.x >> 5;
  int wid  = blockIdx.x * 4 + w;
  int b    = wid / (H_ * (S_ / 16));
  int rem  = wid % (H_ * (S_ / 16));
  int h    = rem / (S_ / 16);
  int q0   = (rem % (S_ / 16)) * 16;

  // bias table for head h lives across the wave: lane i <-> bucket i (32 == 32)
  float bv = rel_bias[lane * H_ + h];
  float biasPos = __shfl(bv, 31);   // rp >= 128 saturated bucket
  float biasNeg = __shfl(bv, 15);   // rp <= -128 saturated bucket

  const bf16* qrow = Q + (size_t)(b * S_ + q0 + l16) * D_ + h * DK_;
  v16bf qa0 = load_a_frag(qrow, half);        // d = 0..31
  v16bf qa1 = load_a_frag(qrow + 32, half);   // d = 32..63

  float m[8], l[8];
  v8f o[4] = {v8f{}, v8f{}, v8f{}, v8f{}};
#pragma unroll
  for (int r = 0; r < 8; ++r) { m[r] = -3.0e38f; l[r] = 0.0f; }

  for (int kc = 0; kc < S_; kc += 32) {
    const bf16* kr0 = Km + (size_t)(b * S_ + kc + l16) * D_ + h * DK_;
    const bf16* kr1 = kr0 + (size_t)16 * D_;
    __builtin_prefetch(kr0 + (size_t)32 * D_, 0, 1);
    __builtin_prefetch(kr1 + (size_t)32 * D_, 0, 1);

    v8f s0 = {}, s1 = {};
    s0 = wmma_bf16(qa0, load_b_frag(kr0, half), s0);
    s0 = wmma_bf16(qa1, load_b_frag(kr0 + 32, half), s0);
    s1 = wmma_bf16(qa0, load_b_frag(kr1, half), s1);
    s1 = wmma_bf16(qa1, load_b_frag(kr1 + 32, half), s1);

    // wave-uniform far-field test: rp spans [kc-(q0+15), kc+31-q0]
    bool farPos = (kc - (q0 + 15)) >= 128;
    bool farNeg = (kc + 31 - q0) <= -128;
    float farBias = farPos ? biasPos : biasNeg;
    bool far = farPos || farNeg;

    float p0[8], p1[8], sc[8];
#pragma unroll
    for (int r = 0; r < 8; ++r) {
      float b0v, b1v;
      if (far) {
        b0v = farBias; b1v = farBias;
      } else {
        int qi = q0 + r + 8 * half;           // C-layout row for this lane half
        b0v = __shfl(bv, t5_bucket((kc + l16) - qi));
        b1v = __shfl(bv, t5_bucket((kc + 16 + l16) - qi));
      }
      float v0 = s0[r] * SM_SCALE + b0v;
      float v1 = s1[r] * SM_SCALE + b1v;
      // row max across the 16 lanes holding this row (xor<=8 stays in-half)
      float rm = fmaxf(v0, v1);
      rm = fmaxf(rm, __shfl_xor(rm, 1));
      rm = fmaxf(rm, __shfl_xor(rm, 2));
      rm = fmaxf(rm, __shfl_xor(rm, 4));
      rm = fmaxf(rm, __shfl_xor(rm, 8));
      float mnew  = fmaxf(m[r], rm);
      float scale = __expf(m[r] - mnew);
      float e0 = __expf(v0 - mnew), e1 = __expf(v1 - mnew);
      float rs = e0 + e1;
      rs += __shfl_xor(rs, 1);
      rs += __shfl_xor(rs, 2);
      rs += __shfl_xor(rs, 4);
      rs += __shfl_xor(rs, 8);
      l[r] = l[r] * scale + rs;
      m[r] = mnew;
      sc[r] = scale;
      p0[r] = e0; p1[r] = e1;
    }
#pragma unroll
    for (int j = 0; j < 4; ++j)
#pragma unroll
      for (int r = 0; r < 8; ++r) o[j][r] *= sc[r];

    // redistribute P: C layout -> LDS row-major 16x32 -> A layout
#pragma unroll
    for (int r = 0; r < 8; ++r) {
      pbuf[w][r + 8 * half][l16]      = (bf16)p0[r];
      pbuf[w][r + 8 * half][16 + l16] = (bf16)p1[r];
    }
    __syncthreads();
    v16bf pa = load_a_frag(&pbuf[w][l16][0], half);
    __syncthreads();

#pragma unroll
    for (int j = 0; j < 4; ++j) {
      const bf16* vr = Vt + (size_t)(h * DK_ + 16 * j + l16) * (B_ * S_)
                          + (size_t)b * S_ + kc;
      __builtin_prefetch(vr + 32, 0, 1);
      o[j] = wmma_bf16(pa, load_b_frag(vr, half), o[j]);
    }
  }

#pragma unroll
  for (int j = 0; j < 4; ++j)
#pragma unroll
    for (int r = 0; r < 8; ++r) {
      int row = q0 + r + 8 * half;
      int col = h * DK_ + 16 * j + l16;
      Oout[(size_t)(b * S_ + row) * D_ + col] = (bf16)(o[j][r] / l[r]);
    }
}

// ---------------- launcher ---------------------------------------------------
extern "C" void kernel_launch(void* const* d_in, const int* in_sizes, int n_in,
                              void* d_out, int out_size, void* d_ws, size_t ws_size,
                              hipStream_t stream) {
  (void)in_sizes; (void)n_in; (void)out_size; (void)ws_size;
  const float* X   = (const float*)d_in[0];
  const float* Wq  = (const float*)d_in[1];
  const float* Wk  = (const float*)d_in[2];
  const float* Wv  = (const float*)d_in[3];
  const float* Wo  = (const float*)d_in[4];
  const float* rel = (const float*)d_in[5];
  float* out = (float*)d_out;

  char*  ws  = (char*)d_ws;
  size_t off = 0;
  auto alloc = [&](size_t bytes) {
    char* p = ws + off;
    off += (bytes + 255) & ~(size_t)255;
    return p;
  };
  const size_t MBS = (size_t)B_ * S_;      // 4096 rows
  bf16* Xb  = (bf16*)alloc(MBS * D_ * 2);
  bf16* WqT = (bf16*)alloc((size_t)D_ * D_ * 2);
  bf16* WkT = (bf16*)alloc((size_t)D_ * D_ * 2);
  bf16* WvT = (bf16*)alloc((size_t)D_ * D_ * 2);
  bf16* WoT = (bf16*)alloc((size_t)D_ * D_ * 2);
  bf16* Qb  = (bf16*)alloc(MBS * D_ * 2);
  bf16* Kb  = (bf16*)alloc(MBS * D_ * 2);
  bf16* Vt  = (bf16*)alloc(MBS * D_ * 2);  // [D_][B_*S_]
  bf16* AO  = (bf16*)alloc(MBS * D_ * 2);

  int nX = (int)(MBS * D_);
  cast_f32_bf16<<<(nX + 255) / 256, 256, 0, stream>>>(X, Xb, nX);
  int nW = D_ * D_;
  dim3 tg((nW + 255) / 256), tb(256);
  transpose_cast<<<tg, tb, 0, stream>>>(Wq, WqT, D_, D_);
  transpose_cast<<<tg, tb, 0, stream>>>(Wk, WkT, D_, D_);
  transpose_cast<<<tg, tb, 0, stream>>>(Wv, WvT, D_, D_);
  transpose_cast<<<tg, tb, 0, stream>>>(Wo, WoT, D_, D_);

  // (M/32)*(N/64) = 128*16 = 2048 waves -> 256 blocks * 8 waves
  dim3 gg(256), gb(256);
  gemm_wmma<<<gg, gb, 0, stream>>>(Xb, WqT, nullptr, Qb, (int)MBS, D_, D_, 1);
  gemm_wmma<<<gg, gb, 0, stream>>>(Xb, WkT, nullptr, Kb, (int)MBS, D_, D_, 1);
  gemm_wmma<<<gg, gb, 0, stream>>>(Xb, WvT, nullptr, Vt, (int)MBS, D_, D_, 2);

  // 4096 (b,h,qtile) waves -> 1024 blocks * 4 waves
  flash_attn_t5<<<1024, 128, 0, stream>>>(Qb, Kb, Vt, rel, AO);

  gemm_wmma<<<gg, gb, 0, stream>>>(AO, WoT, out, nullptr, (int)MBS, D_, D_, 0);
}
